// MultiHeadedSelfAttention_21036749815920
// MI455X (gfx1250) — matrix operand order repaired and checked by reading, not compiled, not run on hardware
//
#include <hip/hip_runtime.h>
#include <hip/hip_bf16.h>

// ---------------- problem constants ----------------
#define B_   2
#define S_   768
#define HID_ 1024
#define NH_  16
#define DK_  64
#define M_   (B_ * S_)          // 1536 rows
#define NEGV (-1.0e18f)
#define INV_SQRT2 0.70710678118f

typedef __bf16 bf16_t;
typedef __attribute__((ext_vector_type(16))) __bf16 v16bf;
typedef __attribute__((ext_vector_type(8)))  float  v8f;

// ---------------- WMMA helpers ----------------
__device__ __forceinline__ v8f wmma_bf16(v16bf a, v16bf b, v8f c) {
  return __builtin_amdgcn_wmma_f32_16x16x32_bf16(
      /*neg_a=*/false, a, /*neg_b=*/false, b,
      /*c_mod=*/(short)0, c, /*reuse_a=*/false, /*reuse_b=*/false);
}

// A-matrix 16x32 bf16 fragment (ISA 7.12.2): lane L<16 holds row L,
// K = {0..7, 16..23}; lane L>=16 holds row L-16, K = {8..15, 24..31}.
__device__ __forceinline__ v16bf load_afrag(const bf16_t* base, int ld) {
  int lane = threadIdx.x & 31;
  int row  = lane & 15;
  int koff = (lane < 16) ? 0 : 8;
  const bf16_t* p = base + (size_t)row * ld + koff;
  union { v16bf v; uint4 u[2]; } r;
  r.u[0] = *reinterpret_cast<const uint4*>(p);
  r.u[1] = *reinterpret_cast<const uint4*>(p + 16);
  return r.v;
}

// B-matrix 32x16 bf16 fragment: column n = lane&15, lanes 0-15 hold
// K={0..7,16..23}, lanes 16-31 hold K={8..15,24..31} (matches A-frag).
// Memory layout: column n of B lives at base + n*ld, contiguous along K.
__device__ __forceinline__ v16bf load_bfrag(const bf16_t* base, int ld) {
  int lane = threadIdx.x & 31;
  int col  = lane & 15;
  int koff = (lane < 16) ? 0 : 8;
  const bf16_t* p = base + (size_t)col * ld + koff;
  union { v16bf v; uint4 u[2]; } r;
  r.u[0] = *reinterpret_cast<const uint4*>(p);
  r.u[1] = *reinterpret_cast<const uint4*>(p + 16);
  return r.v;
}

// reductions within each 16-lane half of the wave (wave32)
__device__ __forceinline__ float half_max(float v) {
  #pragma unroll
  for (int off = 1; off < 16; off <<= 1) v = fmaxf(v, __shfl_xor(v, off, 32));
  return v;
}
__device__ __forceinline__ float half_sum(float v) {
  #pragma unroll
  for (int off = 1; off < 16; off <<= 1) v += __shfl_xor(v, off, 32);
  return v;
}

// ---------------- conversion kernels ----------------
__global__ void cvt_f32_to_bf16(const float* __restrict__ src,
                                bf16_t* __restrict__ dst, int n) {
  int i = blockIdx.x * blockDim.x + threadIdx.x;
  if (i < n) dst[i] = (bf16_t)src[i];
}

// src[R][C] (f32) -> dst[C][R] (bf16)
__global__ void transpose_cvt(const float* __restrict__ src,
                              bf16_t* __restrict__ dst, int R, int C) {
  int i = blockIdx.x * blockDim.x + threadIdx.x;
  if (i >= R * C) return;
  int r = i / C, c = i % C;
  dst[(size_t)c * R + r] = (bf16_t)src[i];
}

// ---------------- rotary kernel ----------------
// q/k layout: [B][NH][S][DK] bf16 (q already pre-scaled by 1/sqrt(DK))
__global__ void rotary_kernel(const bf16_t* __restrict__ qb,
                              const bf16_t* __restrict__ kb,
                              const int* __restrict__ leaf_idx,
                              bf16_t* __restrict__ qr,
                              bf16_t* __restrict__ kr) {
  int t = blockIdx.x * blockDim.x + threadIdx.x;
  const int TOT = B_ * NH_ * S_ * (DK_ / 2);
  if (t >= TOT) return;
  int i   = t & 31;           // freq index 0..31
  int row = t >> 5;           // (b*NH + h)*S + s
  int s   = row % S_;
  int b   = (row / S_) / NH_;
  float li  = (float)leaf_idx[b * S_ + s];
  float inv_freq = __powf(10000.0f, -((float)(2 * i) / (float)DK_));
  float ang = li * inv_freq;
  float sn = __sinf(ang), cs = __cosf(ang);
  size_t base = (size_t)row * DK_ + 2 * i;
  float q1 = (float)qb[base], q2 = (float)qb[base + 1];
  qr[base]     = (bf16_t)(q1 * cs - q2 * sn);
  qr[base + 1] = (bf16_t)(q2 * cs + q1 * sn);
  float k1 = (float)kb[base], k2 = (float)kb[base + 1];
  kr[base]     = (bf16_t)(k1 * cs - k2 * sn);
  kr[base + 1] = (bf16_t)(k2 * cs + k1 * sn);
}

// ---------------- templated WMMA projection GEMM ----------------
// C[M,N] = (A_bf[M,K] @ Wt_bf[N,K]^T + bias) * scale, epilogue by MODE.
// NTILES is compile-time so the inner loop has no branches: accumulators
// stay pinned and WMMAs accumulate in place (no cross-branch v_mov copies).
#define MODE_HEAD  0   // bf16 out -> [B][NH][S][DK]
#define MODE_VT    1   // bf16 out -> [B][NH][DK][S]
#define MODE_GATE  2   // f32 sigmoid(+leaf override) -> [B][NH][S]
#define MODE_OUTF  3   // f32 row-major [M][N]

template <int MODE, int NTILES>
__global__ __launch_bounds__(256) void gemm_proj_kernel(
    const bf16_t* __restrict__ A, const bf16_t* __restrict__ Wt,
    const float* __restrict__ bias, bf16_t* __restrict__ out_bf,
    float* __restrict__ out_f, const int* __restrict__ leaf_idx,
    int M, int N, int K, float scale) {
  int wave = threadIdx.x >> 5;
  int lane = threadIdx.x & 31;
  int m0 = (blockIdx.x * 8 + wave) * 16;
  int n0 = blockIdx.y * (NTILES * 16);

  const bf16_t* arow = A + (size_t)m0 * K;
  v8f acc[NTILES] = {};
  for (int k0 = 0; k0 < K; k0 += 32) {
    v16bf a = load_afrag(arow + k0, K);
    if (k0 + 32 < K)
      __builtin_prefetch(arow + ((size_t)(lane & 15)) * K + k0 + 32, 0, 3);
    #pragma unroll
    for (int t = 0; t < NTILES; ++t) {
      v16bf b = load_bfrag(Wt + (size_t)(n0 + t * 16) * K + k0, K);
      acc[t] = wmma_bf16(a, b, acc[t]);
    }
  }

  int colLane = lane & 15;
  int rbase   = (lane < 16) ? 0 : 8;
  #pragma unroll
  for (int t = 0; t < NTILES; ++t) {
    int ncol = n0 + t * 16 + colLane;
    float bb = bias[ncol];
    #pragma unroll
    for (int r = 0; r < 8; ++r) {
      int m = m0 + rbase + r;
      float val = (acc[t][r] + bb) * scale;
      if (MODE == MODE_HEAD) {
        int b = m / S_, s = m % S_;
        int hh = ncol / DK_, dk = ncol % DK_;
        out_bf[(((size_t)b * NH_ + hh) * S_ + s) * DK_ + dk] = (bf16_t)val;
      } else if (MODE == MODE_VT) {
        int b = m / S_, s = m % S_;
        int hh = ncol / DK_, dk = ncol % DK_;
        out_bf[(((size_t)b * NH_ + hh) * DK_ + dk) * S_ + s] = (bf16_t)val;
      } else if (MODE == MODE_GATE) {
        int b = m / S_, s = m % S_;
        float g = 1.0f / (1.0f + __expf(-val));
        if (leaf_idx[b * S_ + s] == 0) g = 1.0f;
        out_f[((size_t)b * NH_ + ncol) * S_ + s] = g;
      } else {
        out_f[(size_t)m * N + ncol] = val;
      }
    }
  }
}

// ---------------- dual-branch flash attention ----------------
// One wave per 16-row q-tile; both branches (tree & rotary-leaf) maintained
// with online softmax; blend with per-query gate at the end.
__global__ __launch_bounds__(256) void attn_kernel(
    const bf16_t* __restrict__ qb, const bf16_t* __restrict__ kb,
    const bf16_t* __restrict__ qr, const bf16_t* __restrict__ kr,
    const bf16_t* __restrict__ vt, const float* __restrict__ tree,
    const unsigned char* __restrict__ maskp, const int* __restrict__ leaf_idx,
    const float* __restrict__ gate, bf16_t* __restrict__ o_bf) {
  __shared__ bf16_t lds_pt[8][16 * 32];
  __shared__ bf16_t lds_pl[8][16 * 32];

  int wave = threadIdx.x >> 5;
  int lane = threadIdx.x & 31;
  int tile = blockIdx.x * 8 + wave;      // 0 .. B*NH*(S/16)-1
  int qt = tile % (S_ / 16);
  int bh = tile / (S_ / 16);
  int b  = bh / NH_;
  int h  = bh % NH_;
  int q0 = qt * 16;

  const bf16_t* qh  = qb + ((size_t)bh * S_ + q0) * DK_;
  const bf16_t* qrh = qr + ((size_t)bh * S_ + q0) * DK_;
  const bf16_t* kh  = kb + (size_t)bh * S_ * DK_;
  const bf16_t* krh = kr + (size_t)bh * S_ * DK_;
  const bf16_t* vth = vt + (size_t)bh * DK_ * S_;

  v16bf qf0  = load_afrag(qh, DK_);
  v16bf qf1  = load_afrag(qh + 32, DK_);
  v16bf qrf0 = load_afrag(qrh, DK_);
  v16bf qrf1 = load_afrag(qrh + 32, DK_);

  int colLane = lane & 15;
  int rbase   = (lane < 16) ? 0 : 8;

  float gate_r[8];
  bool  lmq[8];
  #pragma unroll
  for (int r = 0; r < 8; ++r) {
    int m = q0 + rbase + r;
    lmq[r]    = (leaf_idx[b * S_ + m] != 0);
    gate_r[r] = gate[(size_t)bh * S_ + m];
  }

  float mt[8], lt[8], ml[8], ll[8];
  #pragma unroll
  for (int r = 0; r < 8; ++r) { mt[r] = ml[r] = -3.0e38f; lt[r] = ll[r] = 0.f; }
  v8f ot[4] = {};
  v8f ol[4] = {};

  for (int k0 = 0; k0 < S_; k0 += 32) {
    // ---- scores: Q K^T for both branches, two 16-key column halves ----
    const bf16_t* kb0 = kh  + (size_t)(k0)      * DK_;
    const bf16_t* kb1 = kh  + (size_t)(k0 + 16) * DK_;
    const bf16_t* kr0 = krh + (size_t)(k0)      * DK_;
    const bf16_t* kr1 = krh + (size_t)(k0 + 16) * DK_;

    v8f st0 = {}, st1 = {}, sl0 = {}, sl1 = {};
    st0 = wmma_bf16(qf0,  load_bfrag(kb0,      DK_), st0);
    st0 = wmma_bf16(qf1,  load_bfrag(kb0 + 32, DK_), st0);
    st1 = wmma_bf16(qf0,  load_bfrag(kb1,      DK_), st1);
    st1 = wmma_bf16(qf1,  load_bfrag(kb1 + 32, DK_), st1);
    sl0 = wmma_bf16(qrf0, load_bfrag(kr0,      DK_), sl0);
    sl0 = wmma_bf16(qrf1, load_bfrag(kr0 + 32, DK_), sl0);
    sl1 = wmma_bf16(qrf0, load_bfrag(kr1,      DK_), sl1);
    sl1 = wmma_bf16(qrf1, load_bfrag(kr1 + 32, DK_), sl1);

    // ---- merge masks + tree_score (streamed once -> non-temporal) ----
    int kc0 = k0 + colLane;
    int kc1 = k0 + 16 + colLane;
    bool pad0 = (maskp[b * S_ + kc0] != 0);
    bool pad1 = (maskp[b * S_ + kc1] != 0);
    bool lk0  = (leaf_idx[b * S_ + kc0] != 0);
    bool lk1  = (leaf_idx[b * S_ + kc1] != 0);
    const float* ts = tree + ((size_t)bh * S_ + q0 + rbase) * S_ + kc0;
    #pragma unroll
    for (int r = 0; r < 8; ++r) {
      float t0 = __builtin_nontemporal_load(ts + (size_t)r * S_);
      float t1 = __builtin_nontemporal_load(ts + (size_t)r * S_ + 16);
      st0[r] = pad0 ? NEGV : (st0[r] + t0) * INV_SQRT2;
      st1[r] = pad1 ? NEGV : (st1[r] + t1) * INV_SQRT2;
      sl0[r] = (lmq[r] && lk0) ? sl0[r] : NEGV;
      sl1[r] = (lmq[r] && lk1) ? sl1[r] : NEGV;
    }

    // ---- online softmax updates (per row, half-wave reductions) ----
    #pragma unroll
    for (int r = 0; r < 8; ++r) {
      float rm = half_max(fmaxf(st0[r], st1[r]));
      float nm = fmaxf(mt[r], rm);
      float sc = __expf(mt[r] - nm);
      float p0 = __expf(st0[r] - nm);
      float p1 = __expf(st1[r] - nm);
      st0[r] = p0; st1[r] = p1;
      lt[r] = lt[r] * sc + half_sum(p0 + p1);
      mt[r] = nm;
      ot[0][r] *= sc; ot[1][r] *= sc; ot[2][r] *= sc; ot[3][r] *= sc;

      rm = half_max(fmaxf(sl0[r], sl1[r]));
      nm = fmaxf(ml[r], rm);
      sc = __expf(ml[r] - nm);
      p0 = __expf(sl0[r] - nm);
      p1 = __expf(sl1[r] - nm);
      sl0[r] = p0; sl1[r] = p1;
      ll[r] = ll[r] * sc + half_sum(p0 + p1);
      ml[r] = nm;
      ol[0][r] *= sc; ol[1][r] *= sc; ol[2][r] *= sc; ol[3][r] *= sc;
    }

    // ---- stage P (D-layout f32) -> LDS (row-major 16x32 bf16) -> A-frag ----
    __syncthreads();
    bf16_t* pt = &lds_pt[wave][0];
    bf16_t* pl = &lds_pl[wave][0];
    #pragma unroll
    for (int r = 0; r < 8; ++r) {
      pt[(rbase + r) * 32 + colLane]      = (bf16_t)st0[r];
      pt[(rbase + r) * 32 + 16 + colLane] = (bf16_t)st1[r];
      pl[(rbase + r) * 32 + colLane]      = (bf16_t)sl0[r];
      pl[(rbase + r) * 32 + 16 + colLane] = (bf16_t)sl1[r];
    }
    __syncthreads();
    v16bf pft = load_afrag(pt, 32);
    v16bf pfl = load_afrag(pl, 32);

    // ---- P @ V : V^T rows are contiguous in key dim -> clean B-frags ----
    #pragma unroll
    for (int t = 0; t < 4; ++t) {
      v16bf vf = load_bfrag(vth + (size_t)(t * 16) * S_ + k0, S_);
      ot[t] = wmma_bf16(pft, vf, ot[t]);
      ol[t] = wmma_bf16(pfl, vf, ol[t]);
    }
  }

  // ---- normalize, blend with gate, store [B][S][HID] bf16 ----
  #pragma unroll
  for (int t = 0; t < 4; ++t) {
    #pragma unroll
    for (int r = 0; r < 8; ++r) {
      int m = q0 + rbase + r;
      float g = gate_r[r];
      float val = g * (ot[t][r] / lt[r]) + (1.0f - g) * (ol[t][r] / ll[r]);
      o_bf[((size_t)(b * S_ + m)) * HID_ + h * DK_ + t * 16 + colLane] = (bf16_t)val;
    }
  }
}

// ---------------- host launcher ----------------
extern "C" void kernel_launch(void* const* d_in, const int* in_sizes, int n_in,
                              void* d_out, int out_size, void* d_ws, size_t ws_size,
                              hipStream_t stream) {
  const float*         content    = (const float*)d_in[0];
  const unsigned char* mask       = (const unsigned char*)d_in[1];
  const int*           leaf_idx   = (const int*)d_in[2];
  const float*         tree_score = (const float*)d_in[3];
  // d_in[4] = leaf_score (unused by the reference math)
  const float* Wq = (const float*)d_in[5];
  const float* bq = (const float*)d_in[6];
  const float* Wk = (const float*)d_in[7];
  const float* bk = (const float*)d_in[8];
  const float* Wv = (const float*)d_in[9];
  const float* bv = (const float*)d_in[10];
  const float* Wo = (const float*)d_in[11];
  const float* bo = (const float*)d_in[12];
  const float* Ws = (const float*)d_in[13];
  const float* bs = (const float*)d_in[14];
  float* out = (float*)d_out;

  // workspace carve-up (256B aligned)
  char*  base = (char*)d_ws;
  size_t off  = 0;
  auto alloc = [&](size_t bytes) {
    char* p = base + off;
    off = (off + bytes + 255) & ~(size_t)255;
    return p;
  };
  bf16_t* c_bf  = (bf16_t*)alloc((size_t)M_ * HID_ * 2);
  bf16_t* wq_t  = (bf16_t*)alloc((size_t)HID_ * HID_ * 2);
  bf16_t* wk_t  = (bf16_t*)alloc((size_t)HID_ * HID_ * 2);
  bf16_t* wv_t  = (bf16_t*)alloc((size_t)HID_ * HID_ * 2);
  bf16_t* wo_t  = (bf16_t*)alloc((size_t)HID_ * HID_ * 2);
  bf16_t* ws_t  = (bf16_t*)alloc((size_t)NH_ * HID_ * 2);
  bf16_t* q_bf  = (bf16_t*)alloc((size_t)M_ * HID_ * 2);
  bf16_t* k_bf  = (bf16_t*)alloc((size_t)M_ * HID_ * 2);
  bf16_t* qr_bf = (bf16_t*)alloc((size_t)M_ * HID_ * 2);
  bf16_t* kr_bf = (bf16_t*)alloc((size_t)M_ * HID_ * 2);
  bf16_t* vt_bf = (bf16_t*)alloc((size_t)M_ * HID_ * 2);
  float*  gatew = (float*) alloc((size_t)B_ * NH_ * S_ * 4);
  bf16_t* o_bf  = (bf16_t*)alloc((size_t)M_ * HID_ * 2);
  (void)ws_size; (void)n_in; (void)in_sizes; (void)out_size;

  const int CT = 256;
  // bf16 conversions (weights transposed: Wt[n][k])
  cvt_f32_to_bf16<<<(M_ * HID_ + CT - 1) / CT, CT, 0, stream>>>(content, c_bf, M_ * HID_);
  transpose_cvt<<<(HID_ * HID_ + CT - 1) / CT, CT, 0, stream>>>(Wq, wq_t, HID_, HID_);
  transpose_cvt<<<(HID_ * HID_ + CT - 1) / CT, CT, 0, stream>>>(Wk, wk_t, HID_, HID_);
  transpose_cvt<<<(HID_ * HID_ + CT - 1) / CT, CT, 0, stream>>>(Wv, wv_t, HID_, HID_);
  transpose_cvt<<<(HID_ * HID_ + CT - 1) / CT, CT, 0, stream>>>(Wo, wo_t, HID_, HID_);
  transpose_cvt<<<(HID_ * NH_  + CT - 1) / CT, CT, 0, stream>>>(Ws, ws_t, HID_, NH_);

  // projections (WMMA GEMMs) — branch-free instantiations
  dim3 gp(M_ / 128, HID_ / 64);
  gemm_proj_kernel<MODE_HEAD, 4><<<gp, 256, 0, stream>>>(
      c_bf, wq_t, bq, q_bf, nullptr, nullptr, M_, HID_, HID_, 0.125f); // /sqrt(64)
  gemm_proj_kernel<MODE_HEAD, 4><<<gp, 256, 0, stream>>>(
      c_bf, wk_t, bk, k_bf, nullptr, nullptr, M_, HID_, HID_, 1.0f);
  gemm_proj_kernel<MODE_VT, 4><<<gp, 256, 0, stream>>>(
      c_bf, wv_t, bv, vt_bf, nullptr, nullptr, M_, HID_, HID_, 1.0f);
  gemm_proj_kernel<MODE_GATE, 1><<<dim3(M_ / 128, 1), 256, 0, stream>>>(
      c_bf, ws_t, bs, nullptr, gatew, leaf_idx, M_, NH_, HID_, 1.0f);

  // rotary embeddings on q (pre-scaled) and k
  int rot_tot = B_ * NH_ * S_ * (DK_ / 2);
  rotary_kernel<<<(rot_tot + CT - 1) / CT, CT, 0, stream>>>(q_bf, k_bf, leaf_idx, qr_bf, kr_bf);

  // dual-branch flash attention: B*NH*(S/16)=1536 q-tiles, 8 waves/block
  attn_kernel<<<(B_ * NH_ * (S_ / 16)) / 8, 256, 0, stream>>>(
      q_bf, k_bf, qr_bf, kr_bf, vt_bf, tree_score, mask, leaf_idx, gatew, o_bf);

  // output projection -> f32 d_out
  gemm_proj_kernel<MODE_OUTF, 4><<<gp, 256, 0, stream>>>(
      o_bf, wo_t, bo, nullptr, out, nullptr, M_, HID_, HID_, 1.0f);
}
